// TopologyDecoder_76811195122386
// MI455X (gfx1250) — matrix operand order repaired and checked by reading, not compiled, not run on hardware
//
#include <hip/hip_runtime.h>
#include <math.h>

typedef __bf16 v16bf __attribute__((ext_vector_type(16)));
typedef float  v8f   __attribute__((ext_vector_type(8)));
typedef float  f4v   __attribute__((ext_vector_type(4)));
typedef int    i4v   __attribute__((ext_vector_type(4)));

#define NB     4096
#define PHID   1024
#define HHID   512
#define MAXP   128
#define MAXHE  42
#define HEPAD  48
#define EMAX   8128

// ---- force global (address_space(1)) memory ops: avoid flat_load/dscnt ----
__device__ __forceinline__ v16bf gload16(const __bf16* p) {
  return *reinterpret_cast<const __attribute__((address_space(1))) v16bf*>(
      (unsigned long long)p);
}
__device__ __forceinline__ void gstore_bf16(__bf16* p, __bf16 v) {
  *reinterpret_cast<__attribute__((address_space(1))) __bf16*>((unsigned long long)p) = v;
}
__device__ __forceinline__ void gprefetch(const __bf16* p) {
  __builtin_prefetch(
      reinterpret_cast<const __attribute__((address_space(1))) char*>((unsigned long long)p), 0, 1);
}

// ---------------------------------------------------------------------------
// A-fragment (16-bit 16x32) slot for element (row, col), K = Kdim.
// Per ISA: lanes 0-15 hold K%32 in {0..7,16..23}; lanes 16-31 in {8..15,24..31}.
// => lane_hi = (c>>3)&1 ; e = (c&7) + ((c>>4)<<3)
// ---------------------------------------------------------------------------
__device__ __forceinline__ size_t apack_index(int row, int col, int Kdim) {
  int rt = row >> 4;
  int kt = col >> 5;
  int c  = col & 31;
  int lanep = (row & 15) + (((c >> 3) & 1) << 4);
  int e = (c & 7) + (((c >> 4) & 1) << 3);
  return ((size_t)(rt * (Kdim >> 5) + kt) * 32 + lanep) * 16 + e;
}

// Pack f32 activations [rows, K] into A-fragment bf16 hi/lo (split for accuracy)
__global__ void pack_a_kernel(const float* __restrict__ X, int K, int total,
                              __bf16* __restrict__ Ah, __bf16* __restrict__ Al) {
  int idx = blockIdx.x * blockDim.x + threadIdx.x;
  if (idx >= total) return;
  int row = idx / K;
  int col = idx % K;
  float v = X[idx];
  __bf16 hi = (__bf16)v;
  __bf16 lo = (__bf16)(v - (float)hi);
  size_t s = apack_index(row, col, K);
  gstore_bf16(Ah + s, hi);
  gstore_bf16(Al + s, lo);
}

// ---------------------------------------------------------------------------
// Pack f32 weight [K, Nw] into WMMA B-fragment bf16 hi/lo order.
// Slot: ((kt*ntiles + nt)*32 + lane)*16 + e
//   K = kt*32 + (lane>=16 ? 8 : 0) + (e&7) + (e>>3)*16 ; N = nt*16 + (lane&15)
// ---------------------------------------------------------------------------
__global__ void pack_w_kernel(const float* __restrict__ W, int K, int Nw, int Npad,
                              __bf16* __restrict__ Wh, __bf16* __restrict__ Wl) {
  int idx = blockIdx.x * blockDim.x + threadIdx.x;
  int total = K * Npad;
  if (idx >= total) return;
  int e      = idx & 15;
  int lane   = (idx >> 4) & 31;
  int ntiles = Npad >> 4;
  int rest   = idx >> 9;
  int nt     = rest % ntiles;
  int kt     = rest / ntiles;
  int kk = kt * 32 + ((lane & 16) ? 8 : 0) + (e & 7) + ((e >> 3) << 4);
  int nn = nt * 16 + (lane & 15);
  float v = (nn < Nw) ? W[(size_t)kk * Nw + nn] : 0.0f;
  __bf16 hi = (__bf16)v;
  __bf16 lo = (__bf16)(v - (float)hi);
  gstore_bf16(Wh + idx, hi);
  gstore_bf16(Wl + idx, lo);
}

// ---------------------------------------------------------------------------
// Core: RT 16-row tiles x NT 16-col tiles per wave. Per K=32 step:
// 2*RT A-fragment loads + 2*NT B-fragment loads feed 3*RT*NT split-bf16 WMMAs
// on RT*NT independent accumulator chains. Pointers advance by constant
// strides (no per-iteration index math).
// ---------------------------------------------------------------------------
template <int RT, int NT>
__device__ __forceinline__ void wmma_multitile(
    const __bf16* __restrict__ Ah, const __bf16* __restrict__ Al,
    const __bf16* __restrict__ Wh, const __bf16* __restrict__ Wl,
    int Kt, int ntiles_n, int rt0, int nts, int lane, v8f acc[RT * NT]) {
#pragma unroll
  for (int q = 0; q < RT * NT; ++q) acc[q] = (v8f){0.f, 0.f, 0.f, 0.f, 0.f, 0.f, 0.f, 0.f};
  const __bf16* aph[RT];
  const __bf16* apl[RT];
#pragma unroll
  for (int i = 0; i < RT; ++i) {
    size_t ab = ((size_t)((rt0 + i) * Kt) * 32 + lane) * 16;
    aph[i] = Ah + ab;
    apl[i] = Al + ab;
  }
  const __bf16* bph[NT];
  const __bf16* bpl[NT];
#pragma unroll
  for (int t = 0; t < NT; ++t) {
    size_t bb = ((size_t)(nts + t) * 32 + lane) * 16;
    bph[t] = Wh + bb;
    bpl[t] = Wl + bb;
  }
  const size_t ASTEP = 32 * 16;                       // one k-tile of one row-tile
  const size_t BSTEP = (size_t)ntiles_n * 32 * 16;    // one k-tile across all n-tiles
  for (int kt = 0; kt < Kt; ++kt) {
    v16bf ah[RT], al[RT];
#pragma unroll
    for (int i = 0; i < RT; ++i) {
      gprefetch(aph[i] + ASTEP);
      ah[i] = gload16(aph[i]);
      al[i] = gload16(apl[i]);
      aph[i] += ASTEP;
      apl[i] += ASTEP;
    }
#pragma unroll
    for (int t = 0; t < NT; ++t) {
      v16bf bh = gload16(bph[t]);
      v16bf bl = gload16(bpl[t]);
      bph[t] += BSTEP;
      bpl[t] += BSTEP;
#pragma unroll
      for (int i = 0; i < RT; ++i) {
        v8f a = acc[i * NT + t];
        a = __builtin_amdgcn_wmma_f32_16x16x32_bf16(false, ah[i], false, bh, (short)0, a, false, false);
        a = __builtin_amdgcn_wmma_f32_16x16x32_bf16(false, ah[i], false, bl, (short)0, a, false, false);
        a = __builtin_amdgcn_wmma_f32_16x16x32_bf16(false, al[i], false, bh, (short)0, a, false, false);
        acc[i * NT + t] = a;
      }
    }
  }
}

// GEMM1: H = gelu(X @ W + b); output stored directly in A-fragment bf16 hi/lo
// order (Hh/Hl) so GEMM2 can consume it without conversion.
template <int RT, int NT>
__global__ void gemm_gelu_kernel(const __bf16* __restrict__ Ah, const __bf16* __restrict__ Al,
                                 const __bf16* __restrict__ Wh, const __bf16* __restrict__ Wl,
                                 const float* __restrict__ bias,
                                 __bf16* __restrict__ Hh, __bf16* __restrict__ Hl,
                                 int K, int N, int totalGroups) {
  int lane = threadIdx.x & 31;
  int wave = threadIdx.x >> 5;
  int grp  = blockIdx.x * (blockDim.x >> 5) + wave;
  if (grp >= totalGroups) return;
  int ntiles_n = N >> 4;
  int ngroups  = ntiles_n / NT;
  int rt0 = (grp / ngroups) * RT;
  int nts = (grp % ngroups) * NT;
  v8f acc[RT * NT];
  wmma_multitile<RT, NT>(Ah, Al, Wh, Wl, K >> 5, ntiles_n, rt0, nts, lane, acc);
  int hs = (lane >> 4) & 1;
#pragma unroll
  for (int i = 0; i < RT; ++i) {
#pragma unroll
    for (int t = 0; t < NT; ++t) {
      int nn = (nts + t) * 16 + (lane & 15);
      float b = bias[nn];
#pragma unroll
      for (int r = 0; r < 8; ++r) {
        int m = ((rt0 + i) << 4) + r + hs * 8;
        float v = acc[i * NT + t][r] + b;
        v = 0.5f * v * (1.0f + erff(v * 0.7071067811865475f));   // exact gelu
        __bf16 hi = (__bf16)v;
        __bf16 lo = (__bf16)(v - (float)hi);
        size_t s = apack_index(m, nn, N);
        gstore_bf16(Hh + s, hi);
        gstore_bf16(Hl + s, lo);
      }
    }
  }
}

// GEMM2 fused with gumbel hard straight-through epilogue (TAU = 1).
template <int RT, int NT>
__global__ void gemm2_gumbel_kernel(const __bf16* __restrict__ Ah, const __bf16* __restrict__ Al,
                                    const __bf16* __restrict__ Wh, const __bf16* __restrict__ Wl,
                                    const float* __restrict__ bias,
                                    const float* __restrict__ g,
                                    int K, int Ncols, int Npad,
                                    float* __restrict__ mask_out, int totalGroups) {
  int lane = threadIdx.x & 31;
  int wave = threadIdx.x >> 5;
  int grp  = blockIdx.x * (blockDim.x >> 5) + wave;
  if (grp >= totalGroups) return;
  int ntiles_n = Npad >> 4;
  int ngroups  = ntiles_n / NT;
  int rt0 = (grp / ngroups) * RT;
  int nts = (grp % ngroups) * NT;
  v8f acc[RT * NT];
  wmma_multitile<RT, NT>(Ah, Al, Wh, Wl, K >> 5, ntiles_n, rt0, nts, lane, acc);
  int hs = (lane >> 4) & 1;
#pragma unroll
  for (int i = 0; i < RT; ++i) {
#pragma unroll
    for (int t = 0; t < NT; ++t) {
      int nn = (nts + t) * 16 + (lane & 15);
      if (nn >= Ncols) continue;
      float b = bias[nn];
#pragma unroll
      for (int r = 0; r < 8; ++r) {
        int m = ((rt0 + i) << 4) + r + hs * 8;
        float l = acc[i * NT + t][r] + b;
        const float* gp = g + ((size_t)m * Ncols + nn) * 2;
        float a0 = l + gp[0];
        float a1 = -l + gp[1];
        float mx = fmaxf(a0, a1);
        float e0 = expf(a0 - mx);
        float e1 = expf(a1 - mx);
        float inv = 1.0f / (e0 + e1);
        float y0 = e0 * inv;
        float y1 = e1 * inv;
        float hard = (y0 > y1) ? 1.0f : 0.0f;
        float outv = (hard + y0) - y0;   // straight-through numerics as in reference
        mask_out[(size_t)m * Ncols + nn] = outv;
      }
    }
  }
}

// Row sum of mask -> int (trunc toward zero, like .astype(int32)); 1 wave / row.
__global__ void row_reduce_kernel(const float* __restrict__ mask, int ldm, int W,
                                  float* __restrict__ out_f, int* __restrict__ out_i) {
  int row  = blockIdx.x;
  int lane = threadIdx.x;
  float s = 0.0f;
  for (int c = lane; c < W; c += 32) s += mask[(size_t)row * ldm + c];
#pragma unroll
  for (int off = 16; off > 0; off >>= 1) s += __shfl_xor(s, off, 32);
  if (lane == 0) {
    int n = (int)s;
    out_i[row] = n;
    out_f[row] = (float)n;
  }
}

// edge_index: lexicographic triu pairs (i<j); rank e = 127*i - i(i-1)/2 + (j-i-1)
__global__ void edge_index_kernel(float* __restrict__ out_ij, int* __restrict__ jj_i) {
  int idx = blockIdx.x * blockDim.x + threadIdx.x;
  if (idx >= MAXP * MAXP) return;
  int i = idx >> 7, j = idx & 127;
  if (i >= j) return;
  int e = 127 * i - (i * (i - 1)) / 2 + (j - i - 1);
  out_ij[e]        = (float)i;
  out_ij[EMAX + e] = (float)j;
  jj_i[e] = j;
}

// edge_valid[b,e] = (jj[e] < n_particles[b]); 133MB streaming -> nontemporal f4 stores
__global__ void edge_valid_kernel(const int* __restrict__ np, const int* __restrict__ jj,
                                  float* __restrict__ out) {
  int b  = blockIdx.y;
  int e4 = blockIdx.x * blockDim.x + threadIdx.x;
  if (e4 >= EMAX / 4) return;
  int n = np[b];
  i4v j4 = *(const i4v*)(jj + 4 * e4);
  f4v v;
#pragma unroll
  for (int t = 0; t < 4; ++t) v[t] = (j4[t] < n) ? 1.0f : 0.0f;
  __builtin_nontemporal_store(v, (f4v*)(out + (size_t)b * EMAX + 4 * e4));
}

// ---------------------------------------------------------------------------
extern "C" void kernel_launch(void* const* d_in, const int* in_sizes, int n_in,
                              void* d_out, int out_size, void* d_ws, size_t ws_size,
                              hipStream_t stream) {
  const float* particle_h  = (const float*)d_in[0];
  // d_in[1] (edge_h) is unused by the reference
  const float* hyperedge_h = (const float*)d_in[2];
  const float* g_p  = (const float*)d_in[3];
  const float* g_h  = (const float*)d_in[4];
  const float* W1_p = (const float*)d_in[5];
  const float* b1_p = (const float*)d_in[6];
  const float* W2_p = (const float*)d_in[7];
  const float* b2_p = (const float*)d_in[8];
  const float* W1_h = (const float*)d_in[9];
  const float* b1_h = (const float*)d_in[10];
  const float* W2_h = (const float*)d_in[11];
  const float* b2_h = (const float*)d_in[12];
  float* out = (float*)d_out;

  // d_out layout (flat, reference return order, all as float)
  const size_t OFF_PM = 0;                                     // [4096,128]
  const size_t OFF_NP = OFF_PM + (size_t)NB * MAXP;            // [4096]
  const size_t OFF_EI = OFF_NP + NB;                           // [2,8128]
  const size_t OFF_EV = OFF_EI + 2 * (size_t)EMAX;             // [4096,8128]
  const size_t OFF_HM = OFF_EV + (size_t)NB * EMAX;            // [4096,42]
  const size_t OFF_NH = OFF_HM + (size_t)NB * MAXHE;           // [4096]

  // workspace carve-up (256B aligned chunks)
  char* ws = (char*)d_ws;
  size_t o = 0;
  auto take = [&](size_t bytes) -> char* {
    char* p = ws + o;
    o += (bytes + 255) & ~(size_t)255;
    return p;
  };
  __bf16* Xp_h = (__bf16*)take((size_t)NB * PHID * 2);
  __bf16* Xp_l = (__bf16*)take((size_t)NB * PHID * 2);
  __bf16* Xh_h = (__bf16*)take((size_t)NB * HHID * 2);
  __bf16* Xh_l = (__bf16*)take((size_t)NB * HHID * 2);
  __bf16* Wp1h = (__bf16*)take((size_t)PHID * PHID * 2);
  __bf16* Wp1l = (__bf16*)take((size_t)PHID * PHID * 2);
  __bf16* Wp2h = (__bf16*)take((size_t)PHID * MAXP * 2);
  __bf16* Wp2l = (__bf16*)take((size_t)PHID * MAXP * 2);
  __bf16* Wh1h = (__bf16*)take((size_t)HHID * HHID * 2);
  __bf16* Wh1l = (__bf16*)take((size_t)HHID * HHID * 2);
  __bf16* Wh2h = (__bf16*)take((size_t)HHID * HEPAD * 2);
  __bf16* Wh2l = (__bf16*)take((size_t)HHID * HEPAD * 2);
  __bf16* Hp_h = (__bf16*)take((size_t)NB * PHID * 2);
  __bf16* Hp_l = (__bf16*)take((size_t)NB * PHID * 2);
  __bf16* Hh_h = (__bf16*)take((size_t)NB * HHID * 2);
  __bf16* Hh_l = (__bf16*)take((size_t)NB * HHID * 2);
  int*    np_i = (int*)take((size_t)NB * 4);
  int*    nh_i = (int*)take((size_t)NB * 4);
  int*    jj_i = (int*)take((size_t)EMAX * 4);

  // 1) pack activations + weights into WMMA fragment order (bf16 hi/lo split)
  pack_a_kernel<<<(NB * PHID + 255) / 256, 256, 0, stream>>>(particle_h, PHID, NB * PHID, Xp_h, Xp_l);
  pack_a_kernel<<<(NB * HHID + 255) / 256, 256, 0, stream>>>(hyperedge_h, HHID, NB * HHID, Xh_h, Xh_l);
  pack_w_kernel<<<(PHID * PHID + 255) / 256, 256, 0, stream>>>(W1_p, PHID, PHID, PHID, Wp1h, Wp1l);
  pack_w_kernel<<<(PHID * MAXP + 255) / 256, 256, 0, stream>>>(W2_p, PHID, MAXP, MAXP, Wp2h, Wp2l);
  pack_w_kernel<<<(HHID * HHID + 255) / 256, 256, 0, stream>>>(W1_h, HHID, HHID, HHID, Wh1h, Wh1l);
  pack_w_kernel<<<(HHID * HEPAD + 255) / 256, 256, 0, stream>>>(W2_h, HHID, MAXHE, HEPAD, Wh2h, Wh2l);

  // 2) hidden layers: GEMM + exact gelu, output directly in A-fragment order
  {
    int groups = ((NB / 16) / 2) * ((PHID / 16) / 4);   // 128 * 16 = 2048 wave-jobs
    gemm_gelu_kernel<2, 4><<<(groups + 7) / 8, 256, 0, stream>>>(Xp_h, Xp_l, Wp1h, Wp1l, b1_p,
                                                                 Hp_h, Hp_l, PHID, PHID, groups);
  }
  {
    int groups = ((NB / 16) / 2) * ((HHID / 16) / 4);   // 128 * 8 = 1024
    gemm_gelu_kernel<2, 4><<<(groups + 7) / 8, 256, 0, stream>>>(Xh_h, Xh_l, Wh1h, Wh1l, b1_h,
                                                                 Hh_h, Hh_l, HHID, HHID, groups);
  }

  // 3) logits + gumbel hard-ST masks
  {
    int groups = ((NB / 16) / 2) * ((MAXP / 16) / 4);   // 128 * 2 = 256
    gemm2_gumbel_kernel<2, 4><<<(groups + 7) / 8, 256, 0, stream>>>(Hp_h, Hp_l, Wp2h, Wp2l, b2_p, g_p,
                                                                    PHID, MAXP, MAXP,
                                                                    out + OFF_PM, groups);
  }
  {
    int groups = ((NB / 16) / 2) * ((HEPAD / 16) / 3);  // 128 * 1 = 128
    gemm2_gumbel_kernel<2, 3><<<(groups + 7) / 8, 256, 0, stream>>>(Hh_h, Hh_l, Wh2h, Wh2l, b2_h, g_h,
                                                                    HHID, MAXHE, HEPAD,
                                                                    out + OFF_HM, groups);
  }

  // 4) n_particles / n_hyperedges (row sums, truncation like astype(int32))
  row_reduce_kernel<<<NB, 32, 0, stream>>>(out + OFF_PM, MAXP, MAXP, out + OFF_NP, np_i);
  row_reduce_kernel<<<NB, 32, 0, stream>>>(out + OFF_HM, MAXHE, MAXHE, out + OFF_NH, nh_i);

  // 5) edge_index (constant triu pairs) + jj table
  edge_index_kernel<<<(MAXP * MAXP + 255) / 256, 256, 0, stream>>>(out + OFF_EI, jj_i);

  // 6) edge_valid: 4096 x 8128 streaming writes, nontemporal
  {
    dim3 grid((EMAX / 4 + 255) / 256, NB);
    edge_valid_kernel<<<grid, 256, 0, stream>>>(np_i, jj_i, out + OFF_EV);
  }
}
